// Model_74062416052792
// MI455X (gfx1250) — matrix operand order repaired and checked by reading, not compiled, not run on hardware
//
#include <hip/hip_runtime.h>
#include <hip/hip_bf16.h>
#include <math.h>

typedef __attribute__((ext_vector_type(16))) __bf16 v16bf;
typedef __attribute__((ext_vector_type(8)))  __bf16 v8bf;
typedef __attribute__((ext_vector_type(8)))  float  v8f;
typedef __attribute__((ext_vector_type(4)))  int    v4i;

#define T_STEPS 512
#define BATCH   64
#define HDIM    128
#define IDIM    2048

#if __has_builtin(__builtin_amdgcn_global_load_async_to_lds_b128)
#define USE_ASYNC_LDS 1
typedef __attribute__((address_space(1))) v4i gv4i;   // global b128 chunk
typedef __attribute__((address_space(3))) v4i lv4i;   // LDS b128 chunk
#else
#define USE_ASYNC_LDS 0
#endif

static __device__ __forceinline__ v8f wmma_bf16(v16bf a, v16bf b, v8f c) {
  return __builtin_amdgcn_wmma_f32_16x16x32_bf16(false, a, false, b, (short)0, c,
                                                 false, false);
}

// Stage one pre-packed B chunk (256 lanes x 16 bf16 = 8 KB) into LDS.
static __device__ __forceinline__ void stage_b(const __bf16* wpack, __bf16* dst,
                                               int c, int tid) {
  const __bf16* src = wpack + ((size_t)c * 256 + tid) * 16;
#if USE_ASYNC_LDS
  __builtin_amdgcn_global_load_async_to_lds_b128(
      (gv4i*)const_cast<__bf16*>(src),
      (lv4i*)(dst + tid * 16), 0, 0);
  __builtin_amdgcn_global_load_async_to_lds_b128(
      (gv4i*)const_cast<__bf16*>(src + 8),
      (lv4i*)(dst + tid * 16 + 8), 0, 0);
#else
  *(v16bf*)(dst + tid * 16) = *(const v16bf*)src;
#endif
}

// ---------------------------------------------------------------------------
// Kernel 1: pack W_ih (fp32 [128 x 2048]) into bf16 WMMA B-fragment layout.
// Flat index: ((c*8 + nt)*32 + L)*16 + e  <-  W[(nt*16+L%16)*IDIM + c*32 + ((e&8)<<1) + (L>>4)*8 + (e&7)]
// ---------------------------------------------------------------------------
__global__ __launch_bounds__(256) void pack_wih(const float* __restrict__ W,
                                                __bf16* __restrict__ dst) {
  int flat = blockIdx.x * 256 + threadIdx.x;   // covers 128*2048 = 262144
  int e  = flat & 15;
  int L  = (flat >> 4) & 31;
  int nt = (flat >> 9) & 7;
  int c  = flat >> 12;
  dst[flat] = (__bf16)W[(nt * 16 + (L & 15)) * IDIM + c * 32 + ((e & 8) << 1) + (L >> 4) * 8 + (e & 7)];
}

// ---------------------------------------------------------------------------
// Kernel 2: xp[t, b, n] = sum_k x[b,t,k] * W_ih[n,k] + b_ih[n]   (t-major out)
//   M = 32768 (B*T), K = 2048, N = 128.  128x128 tile/WG, 8 waves.
//   A: register-pipelined fp32->bf16 LDS staging (latency hidden behind
//   barrier + WMMA).  B: async global->LDS copy (ASYNCcnt) when available.
// ---------------------------------------------------------------------------
__global__ __launch_bounds__(256) void gemm_xp(const float* __restrict__ x,
                                               const __bf16* __restrict__ wpack,
                                               const float* __restrict__ b_ih,
                                               float* __restrict__ xp) {
  __shared__ __bf16 a_lds[2][128 * 32];  // 2 x 8 KB
  __shared__ __bf16 b_lds[2][256 * 16];  // 2 x 8 KB

  const int tid  = threadIdx.x;
  const int wave = tid >> 5;
  const int L    = tid & 31;
  const int m0   = blockIdx.x * 128;
  const int row  = tid >> 3;          // 0..31
  const int col  = (tid & 7) * 4;     // 0,4,..,28

  v8f acc[8] = {};

  // prologue: chunk 0 A tile in registers, chunk 0 B staged (async) to LDS
  float4 r[4];
  #pragma unroll
  for (int rr = 0; rr < 4; ++rr)
    r[rr] = *(const float4*)(x + (size_t)(m0 + row + rr * 32) * IDIM + col);
  stage_b(wpack, &b_lds[0][0], 0, tid);

  for (int c = 0; c < 64; ++c) {
    const int buf = c & 1;

    // commit A regs (chunk c) to LDS as bf16
    #pragma unroll
    for (int rr = 0; rr < 4; ++rr) {
      __bf16* d = &a_lds[buf][(row + rr * 32) * 32 + col];
      d[0] = (__bf16)r[rr].x; d[1] = (__bf16)r[rr].y;
      d[2] = (__bf16)r[rr].z; d[3] = (__bf16)r[rr].w;
    }
    // prefetch next A chunk into regs (clamped; redundant last iter)
    const int cn = c < 63 ? c + 1 : 63;
    #pragma unroll
    for (int rr = 0; rr < 4; ++rr)
      r[rr] = *(const float4*)(x + (size_t)(m0 + row + rr * 32) * IDIM +
                               cn * 32 + col);

#if USE_ASYNC_LDS
    asm volatile("s_wait_asynccnt 0x0" ::: "memory");  // B(c) landed in LDS
#endif
    __syncthreads();  // staging(c) visible AND compute(c-1) finished

    // issue async B copy for next chunk into the other buffer
    if (c < 63) stage_b(wpack, &b_lds[buf ^ 1][0], c + 1, tid);

    // A fragment for this wave: two 8-element K runs per lane
    const int arow = wave * 16 + (L & 15);
    const int kg   = (L >> 4) * 8;
    v8bf lo = *(const v8bf*)(&a_lds[buf][arow * 32 + kg]);
    v8bf hi = *(const v8bf*)(&a_lds[buf][arow * 32 + 16 + kg]);
    v16bf af;
    #pragma unroll
    for (int i = 0; i < 8; ++i) { af[i] = lo[i]; af[i + 8] = hi[i]; }

    #pragma unroll
    for (int nt = 0; nt < 8; ++nt) {
      v16bf bfrag = *(const v16bf*)(&b_lds[buf][(nt * 32 + L) * 16]);
      acc[nt] = wmma_bf16(af, bfrag, acc[nt]);
    }
  }

  // epilogue: bias add + fp32 store, t-major layout xp[t][b][n]
  const int nloc  = L & 15;
  const int mhalf = (L >> 4) * 8;
  #pragma unroll
  for (int nt = 0; nt < 8; ++nt) {
    const int n    = nt * 16 + nloc;
    const float bb = b_ih[n];
    const int mb   = m0 + wave * 16 + mhalf;
    #pragma unroll
    for (int v = 0; v < 8; ++v) {
      const int m  = mb + v;
      const int bi = m >> 9;      // batch index
      const int tt = m & 511;     // timestep
      xp[((size_t)tt * BATCH + bi) * HDIM + n] = acc[nt][v] + bb;
    }
  }
}

// ---------------------------------------------------------------------------
// Kernel 3: recurrence + FFN + LayerNorm.
//   4 WGs x 256 threads; WG g owns batch rows [g*16, g*16+16).
//   W_hh B-fragments hoisted to registers; h double-buffered in LDS ->
//   exactly ONE barrier per step; xp[t+1] prefetched during step t.
// ---------------------------------------------------------------------------
__global__ __launch_bounds__(256) void rnn_ffn(const float* __restrict__ xp,
                                               const float* __restrict__ W_hh,
                                               const float* __restrict__ b_hh,
                                               const float* __restrict__ W1,
                                               const float* __restrict__ b1,
                                               const float* __restrict__ W2,
                                               const float* __restrict__ b2,
                                               const float* __restrict__ gamma,
                                               const float* __restrict__ beta,
                                               float* __restrict__ out) {
  __shared__ __bf16 Bw[128 * 128];       // packed weight, 32 KB
  __shared__ __bf16 h_lds[2][16 * 128];  // double-buffered hidden state, 8 KB
  __shared__ float  zf[16 * 128];        // fp32 staging for layernorm, 8 KB
  __shared__ float  mu_s[16], rs_s[16];

  const int tid = threadIdx.x;
  const int nt  = tid >> 5;           // wave id == its N tile
  const int L   = tid & 31;
  const int b0  = blockIdx.x * 16;

  // pack W_hh into LDS B-layout; zero h0 (buffer 0)
  #pragma unroll 4
  for (int j = 0; j < 64; ++j) {
    int flat = j * 256 + tid;
    int e = flat & 15, Lx = (flat >> 4) & 31, nx = (flat >> 9) & 7, c = flat >> 12;
    Bw[flat] = (__bf16)W_hh[(nx * 16 + (Lx & 15)) * HDIM + c * 32 + ((e & 8) << 1) + (Lx >> 4) * 8 + (e & 7)];
  }
  #pragma unroll
  for (int j = 0; j < 8; ++j) h_lds[0][j * 256 + tid] = (__bf16)0.0f;
  __syncthreads();

  const int mloc  = L & 15;
  const int kg    = (L >> 4) * 8;
  const int mhalf = (L >> 4) * 8;
  const int n     = nt * 16 + mloc;
  const float bhh_n = b_hh[n];

  // hoist the loop-invariant W_hh B fragments into registers
  v16bf bw[4];
  #pragma unroll
  for (int c = 0; c < 4; ++c)
    bw[c] = *(const v16bf*)(&Bw[(c * 8 + nt) * 512 + L * 16]);

  // prefetch xp for t = 0 (xp is t-major: [t][b][n])
  float xv[8];
  #pragma unroll
  for (int v = 0; v < 8; ++v)
    xv[v] = xp[((size_t)0 * BATCH + b0 + mhalf + v) * HDIM + n];

  for (int t = 0; t < T_STEPS; ++t) {
    const int rb = t & 1;       // read buffer (h_{t-1})
    const int wb = rb ^ 1;      // write buffer (h_t)

    v16bf a4[4];
    #pragma unroll
    for (int c = 0; c < 4; ++c) {
      v8bf lo = *(const v8bf*)(&h_lds[rb][mloc * 128 + c * 32 + kg]);
      v8bf hi = *(const v8bf*)(&h_lds[rb][mloc * 128 + c * 32 + 16 + kg]);
      #pragma unroll
      for (int i = 0; i < 8; ++i) { a4[c][i] = lo[i]; a4[c][8 + i] = hi[i]; }
    }

    v8f acc = {};
    #pragma unroll
    for (int c = 0; c < 4; ++c) acc = wmma_bf16(a4[c], bw[c], acc);

    // prefetch next timestep's activations (dense 32 KB block, clamped)
    const int tn = t + 1 < T_STEPS ? t + 1 : t;
    float xn[8];
    #pragma unroll
    for (int v = 0; v < 8; ++v)
      xn[v] = xp[((size_t)tn * BATCH + b0 + mhalf + v) * HDIM + n];

    #pragma unroll
    for (int v = 0; v < 8; ++v) {
      const int m = mhalf + v;
      float hv = tanhf(acc[v] + xv[v] + bhh_n);
      h_lds[wb][m * 128 + n] = (__bf16)hv;
    }
    __syncthreads();            // single barrier per step (double-buffered h)

    #pragma unroll
    for (int v = 0; v < 8; ++v) xv[v] = xn[v];
  }
  // t = 511 wrote buffer 0 -> hT lives in h_lds[0]

  // ---- FC1: z = relu(hT @ W1^T + b1) : read h_lds[0], write h_lds[1] ----
  #pragma unroll 4
  for (int j = 0; j < 64; ++j) {
    int flat = j * 256 + tid;
    int e = flat & 15, Lx = (flat >> 4) & 31, nx = (flat >> 9) & 7, c = flat >> 12;
    Bw[flat] = (__bf16)W1[(nx * 16 + (Lx & 15)) * HDIM + c * 32 + ((e & 8) << 1) + (Lx >> 4) * 8 + (e & 7)];
  }
  __syncthreads();
  {
    v8f acc = {};
    #pragma unroll
    for (int c = 0; c < 4; ++c) {
      v8bf lo = *(const v8bf*)(&h_lds[0][mloc * 128 + c * 32 + kg]);
      v8bf hi = *(const v8bf*)(&h_lds[0][mloc * 128 + c * 32 + 16 + kg]);
      v16bf af;
      #pragma unroll
      for (int i = 0; i < 8; ++i) { af[i] = lo[i]; af[i + 8] = hi[i]; }
      v16bf bfrag = *(const v16bf*)(&Bw[(c * 8 + nt) * 512 + L * 16]);
      acc = wmma_bf16(af, bfrag, acc);
    }
    const float b1n = b1[n];
    #pragma unroll
    for (int v = 0; v < 8; ++v) {
      const int m = mhalf + v;
      float z = acc[v] + b1n;
      z = z > 0.0f ? z : 0.0f;
      h_lds[1][m * 128 + n] = (__bf16)z;
    }
    __syncthreads();
  }

  // ---- FC2: z = relu(z @ W2^T + b2) : read h_lds[1], fp32 out to zf ----
  #pragma unroll 4
  for (int j = 0; j < 64; ++j) {
    int flat = j * 256 + tid;
    int e = flat & 15, Lx = (flat >> 4) & 31, nx = (flat >> 9) & 7, c = flat >> 12;
    Bw[flat] = (__bf16)W2[(nx * 16 + (Lx & 15)) * HDIM + c * 32 + ((e & 8) << 1) + (Lx >> 4) * 8 + (e & 7)];
  }
  __syncthreads();
  {
    v8f acc = {};
    #pragma unroll
    for (int c = 0; c < 4; ++c) {
      v8bf lo = *(const v8bf*)(&h_lds[1][mloc * 128 + c * 32 + kg]);
      v8bf hi = *(const v8bf*)(&h_lds[1][mloc * 128 + c * 32 + 16 + kg]);
      v16bf af;
      #pragma unroll
      for (int i = 0; i < 8; ++i) { af[i] = lo[i]; af[i + 8] = hi[i]; }
      v16bf bfrag = *(const v16bf*)(&Bw[(c * 8 + nt) * 512 + L * 16]);
      acc = wmma_bf16(af, bfrag, acc);
    }
    const float b2n = b2[n];
    #pragma unroll
    for (int v = 0; v < 8; ++v) {
      const int m = mhalf + v;
      float z = acc[v] + b2n;
      z = z > 0.0f ? z : 0.0f;
      zf[m * 128 + n] = z;
    }
    __syncthreads();
  }

  // ---- LayerNorm over 128 features of each of the 16 rows ----
  if (tid < 16) {
    float s = 0.0f;
    for (int j = 0; j < 128; ++j) s += zf[tid * 128 + j];
    const float mu = s * (1.0f / 128.0f);
    float var = 0.0f;
    for (int j = 0; j < 128; ++j) {
      float d = zf[tid * 128 + j] - mu;
      var += d * d;
    }
    mu_s[tid] = mu;
    rs_s[tid] = rsqrtf(var * (1.0f / 128.0f) + 1e-5f);
  }
  __syncthreads();
  #pragma unroll
  for (int j = 0; j < 8; ++j) {
    const int idx = tid * 8 + j;           // 0..2047
    const int m = idx >> 7, nn = idx & 127;
    out[(size_t)(b0 + m) * HDIM + nn] =
        gamma[nn] * (zf[idx] - mu_s[m]) * rs_s[m] + beta[nn];
  }
}

// ---------------------------------------------------------------------------
extern "C" void kernel_launch(void* const* d_in, const int* in_sizes, int n_in,
                              void* d_out, int out_size, void* d_ws, size_t ws_size,
                              hipStream_t stream) {
  const float* x     = (const float*)d_in[0];
  const float* W_ih  = (const float*)d_in[1];
  const float* b_ih  = (const float*)d_in[2];
  const float* W_hh  = (const float*)d_in[3];
  const float* b_hh  = (const float*)d_in[4];
  const float* W1    = (const float*)d_in[5];
  const float* b1    = (const float*)d_in[6];
  const float* W2    = (const float*)d_in[7];
  const float* b2    = (const float*)d_in[8];
  const float* gamma = (const float*)d_in[9];
  const float* beta  = (const float*)d_in[10];
  float* out = (float*)d_out;

  __bf16* wpack = (__bf16*)d_ws;                      // 128*2048 bf16 = 512 KB
  float*  xp    = (float*)((char*)d_ws + (1u << 20)); // 512*64*128 fp32 = 16 MB

  pack_wih<<<1024, 256, 0, stream>>>(W_ih, wpack);
  gemm_xp <<<256,  256, 0, stream>>>(x, wpack, b_ih, xp);
  rnn_ffn <<<4,    256, 0, stream>>>(xp, W_hh, b_hh, W1, b1, W2, b2,
                                     gamma, beta, out);
}
